// WideAndDeep_25074019074517
// MI455X (gfx1250) — compile-verified, matrix operand order checked by reading, not run on hardware
//
#include <hip/hip_runtime.h>
#include <hip/hip_bf16.h>

// CDNA5 / gfx1250: wave32, WMMA 16x16x32 bf16, async-to-LDS staging
typedef __attribute__((ext_vector_type(16))) __bf16 v16bf;
typedef __attribute__((ext_vector_type(8)))  float  v8f;

constexpr int Bn = 8192, Ln = 64, Dn = 512, DKn = 64, Tn = 128;
constexpr int WIDE_IN = 2 * Tn + Tn * Tn;   // 16640
constexpr int DEEP_IN = 3 * Dn;             // 1536

__device__ __forceinline__ float sigf(float x) { return 1.0f / (1.0f + __expf(-x)); }

// ---------------------------------------------------------------- fp32 -> bf16
__global__ void wd_cvt_bf16(const float* __restrict__ s, __bf16* __restrict__ d, int n) {
    int i = blockIdx.x * blockDim.x + threadIdx.x;
    if (i < n) d[i] = (__bf16)s[i];
}

// ------------------------------------------- pack deep_in slices 0 and 2 (bf16)
__global__ void wd_pack_deep_in(const float* __restrict__ mash,
                                const float* __restrict__ cand,
                                __bf16* __restrict__ din) {
    int i = blockIdx.x * blockDim.x + threadIdx.x;            // [0, B*512)
    if (i >= Bn * Dn) return;
    int b = i >> 9, d = i & 511;
    size_t row = (size_t)b * DEEP_IN;
    din[row + d]            = (__bf16)mash[i];
    din[row + 2 * Dn + d]   = (__bf16)cand[i];
}

// ------------------------------------------------------- wide branch GEMV (HBM-bound)
__global__ void wd_wide(const float* __restrict__ mash, const float* __restrict__ targ,
                        const float* __restrict__ cross, const float* __restrict__ w,
                        const float* __restrict__ wb, float* __restrict__ wide_lin) {
    int b = blockIdx.x, tid = threadIdx.x;
    float acc = 0.f;
    for (int i = tid; i < WIDE_IN; i += 256) {
        float x;
        if (i < Tn)            x = mash[(size_t)b * Tn + i];
        else if (i < 2 * Tn)   x = targ[(size_t)b * Tn + i - Tn];
        else                   x = cross[(size_t)b * (Tn * Tn) + i - 2 * Tn];
        acc += x * w[i];
    }
    for (int o = 16; o > 0; o >>= 1) acc += __shfl_xor(acc, o, 32);
    __shared__ float part[8];
    if ((tid & 31) == 0) part[tid >> 5] = acc;
    __syncthreads();
    if (tid == 0) {
        float s = 0.f;
        for (int i = 0; i < 8; ++i) s += part[i];
        wide_lin[b] = s + wb[0];
    }
}

// ---------------- fused ragged attention: qk -> scores -> masked softmax -> context
// One block (256 thr) per batch row. E[b] (128 KB) is staged into LDS with
// GLOBAL_LOAD_ASYNC_TO_LDS_B128 (ASYNCcnt path) so the DMA overlaps the Q/qk
// compute that does not depend on E; E is read from HBM exactly once.
__global__ void wd_attn(const float* __restrict__ E, const float* __restrict__ cand,
                        const int* __restrict__ clen,
                        const float* __restrict__ WK, const float* __restrict__ WQ,
                        __bf16* __restrict__ din) {
    extern __shared__ float sm[];
    float* Es  = sm;             // 64*512 = 32768
    float* qks = sm + 32768;     // 512
    float* Cs  = sm + 33280;     // 512
    float* Qs  = sm + 33792;     // 64  (Q, then reused for scores)
    float* Pr  = sm + 33856;     // 64  (probs)

    int b = blockIdx.x, tid = threadIdx.x, lane = tid & 31, wv = tid >> 5;

    // --- issue async E staging (each thread: 32 x 16B chunks) ---
    {
        unsigned lbase = (unsigned)(uintptr_t)Es;
        uint64_t gbase = (uint64_t)(uintptr_t)(E + (size_t)b * Ln * Dn);
        for (int i = tid; i < (Ln * Dn) / 4; i += 256) {
            unsigned la = lbase + (unsigned)i * 16u;
            uint64_t ga = gbase + (uint64_t)i * 16u;
            asm volatile("global_load_async_to_lds_b128 %0, %1, off"
                         :: "v"(la), "v"(ga) : "memory");
        }
    }

    // --- work that does not need E: candidate vector, Q, qk ---
    for (int i = tid; i < Dn; i += 256) Cs[i] = cand[(size_t)b * Dn + i];
    __syncthreads();

    // Q[k] = c . W_Q[k,:]   (8 waves x 8 rows, lane-strided + shuffle reduce)
    for (int r = 0; r < 8; ++r) {
        int k = wv * 8 + r;
        float p = 0.f;
        for (int j = 0; j < 16; ++j) { int d = lane + 32 * j; p += Cs[d] * WQ[k * Dn + d]; }
        for (int o = 16; o > 0; o >>= 1) p += __shfl_xor(p, o, 32);
        if (lane == 0) Qs[k] = p;
    }
    __syncthreads();

    // qk[d] = sum_k Q[k] * W_K[k,d]   (coalesced in d)
    for (int d = tid; d < Dn; d += 256) {
        float s = 0.f;
        for (int k = 0; k < DKn; ++k) s += Qs[k] * WK[k * Dn + d];
        qks[d] = s;
    }

    // drain this wave's async LDS writes, then barrier: E + qk now visible to all
    asm volatile("s_wait_asynccnt 0" ::: "memory");
    __syncthreads();

    // scores[l] = E[l,:] . qk   (reuse Qs)
    float sc[8];
    for (int r = 0; r < 8; ++r) {
        int l = wv * 8 + r;
        float p = 0.f;
        for (int j = 0; j < 16; ++j) { int d = lane + 32 * j; p += Es[l * Dn + d] * qks[d]; }
        for (int o = 16; o > 0; o >>= 1) p += __shfl_xor(p, o, 32);
        sc[r] = p;
    }
    __syncthreads();
    for (int r = 0; r < 8; ++r) if (lane == 0) Qs[wv * 8 + r] = sc[r];
    __syncthreads();

    // masked softmax over L=64 (wave 0); len==0 -> all-zero probs (matches reference)
    int len = clen[b];
    if (tid < 32) {
        float s0 = Qs[lane], s1 = Qs[lane + 32];
        float v0 = (lane < len) ? s0 : -1e30f;
        float v1 = (lane + 32 < len) ? s1 : -1e30f;
        float m = fmaxf(v0, v1);
        for (int o = 16; o > 0; o >>= 1) m = fmaxf(m, __shfl_xor(m, o, 32));
        float e0 = (lane < len) ? __expf(s0 - m) : 0.f;
        float e1 = (lane + 32 < len) ? __expf(s1 - m) : 0.f;
        float s = e0 + e1;
        for (int o = 16; o > 0; o >>= 1) s += __shfl_xor(s, o, 32);
        float inv = 1.f / fmaxf(s, 1e-30f);
        Pr[lane] = e0 * inv;
        Pr[lane + 32] = e1 * inv;
    }
    __syncthreads();

    // api_context[d] = sum_l probs[l] * E[l,d]  -> bf16 middle slice of deep_in
    for (int d = tid; d < Dn; d += 256) {
        float a = 0.f;
        for (int l = 0; l < Ln; ++l) a += Pr[l] * Es[l * Dn + d];
        din[(size_t)b * DEEP_IN + Dn + d] = (__bf16)a;
    }
}

// -------- bf16 WMMA GEMM: C = act(A(MxK) * W(NxK)^T + bias), bf16 out
// Each wave computes a 16(M) x 64(N) strip with 4 accumulators; A fragment is
// reused across 4 WMMAs. The k-loop is manually double-buffered so the next
// step's 5 fragment loads are in flight while the current WMMAs execute
// (avoids full s_wait_loadcnt drains). Operands are L2-resident
// (deep_in 25MB, weights <2MB << 192MB global L2).
__global__ void wd_gemm_bf16(const __bf16* __restrict__ A, const __bf16* __restrict__ W,
                             const float* __restrict__ bias, __bf16* __restrict__ C,
                             int M, int N, int K, int relu) {
    int wid  = (blockIdx.x * blockDim.x + threadIdx.x) >> 5;
    int lane = threadIdx.x & 31;
    int tilesN = N >> 6;                       // strips of 64 columns
    int tM = (wid / tilesN) << 4;
    int tN = (wid % tilesN) << 6;
    if (tM >= M) return;
    int half = lane >> 4, m = lane & 15;

    const __bf16* Arow = A + (size_t)(tM + m) * K + half * 16;
    const __bf16* W0 = W + (size_t)(tN + m) * K + half * 16;
    size_t ws1 = (size_t)16 * K, ws2 = 2 * ws1, ws3 = 3 * ws1;

    v8f acc0 = {}, acc1 = {}, acc2 = {}, acc3 = {};

    // prologue: fragments for k = 0
    v16bf a  = *(const v16bf*)(Arow);
    v16bf b0 = *(const v16bf*)(W0);
    v16bf b1 = *(const v16bf*)(W0 + ws1);
    v16bf b2 = *(const v16bf*)(W0 + ws2);
    v16bf b3 = *(const v16bf*)(W0 + ws3);

    for (int k = 32; k < K; k += 32) {
        __builtin_prefetch(Arow + k + 128, 0, 0);   // global_prefetch_b8
        // issue next-step loads before consuming current fragments
        v16bf an  = *(const v16bf*)(Arow + k);
        v16bf bn0 = *(const v16bf*)(W0 + k);
        v16bf bn1 = *(const v16bf*)(W0 + ws1 + k);
        v16bf bn2 = *(const v16bf*)(W0 + ws2 + k);
        v16bf bn3 = *(const v16bf*)(W0 + ws3 + k);
        acc0 = __builtin_amdgcn_wmma_f32_16x16x32_bf16(false, a, false, b0, (short)0, acc0, false, false);
        acc1 = __builtin_amdgcn_wmma_f32_16x16x32_bf16(false, a, false, b1, (short)0, acc1, false, false);
        acc2 = __builtin_amdgcn_wmma_f32_16x16x32_bf16(false, a, false, b2, (short)0, acc2, false, false);
        acc3 = __builtin_amdgcn_wmma_f32_16x16x32_bf16(false, a, false, b3, (short)0, acc3, false, false);
        a = an; b0 = bn0; b1 = bn1; b2 = bn2; b3 = bn3;
    }
    // epilogue WMMAs for the last k-step
    acc0 = __builtin_amdgcn_wmma_f32_16x16x32_bf16(false, a, false, b0, (short)0, acc0, false, false);
    acc1 = __builtin_amdgcn_wmma_f32_16x16x32_bf16(false, a, false, b1, (short)0, acc1, false, false);
    acc2 = __builtin_amdgcn_wmma_f32_16x16x32_bf16(false, a, false, b2, (short)0, acc2, false, false);
    acc3 = __builtin_amdgcn_wmma_f32_16x16x32_bf16(false, a, false, b3, (short)0, acc3, false, false);

    int rbase = tM + half * 8;
    v8f accs[4] = {acc0, acc1, acc2, acc3};
    for (int g = 0; g < 4; ++g) {
        int col = tN + g * 16 + m;
        float bz = bias[col];
        for (int r = 0; r < 8; ++r) {
            float f = accs[g][r] + bz;
            if (relu) f = fmaxf(f, 0.f);
            C[(size_t)(rbase + r) * N + col] = (__bf16)f;
        }
    }
}

// ---------------------------- layer4 GEMV + wide/deep combine + final sigmoid
__global__ void wd_final(const __bf16* __restrict__ h3, const __bf16* __restrict__ w4,
                         const float* __restrict__ d4b, const float* __restrict__ wide_lin,
                         const float* __restrict__ ww, const float* __restrict__ dw,
                         const float* __restrict__ bias0, float* __restrict__ out) {
    int i = blockIdx.x * blockDim.x + threadIdx.x;
    if (i >= Bn) return;
    const __bf16* h = h3 + (size_t)i * 128;
    float a = 0.f;
    for (int j = 0; j < 128; ++j) a += (float)h[j] * (float)w4[j];
    a += d4b[0];
    float o = ww[0] * sigf(wide_lin[i]) + dw[0] * a + bias0[0];
    out[i] = sigf(o);
}

extern "C" void kernel_launch(void* const* d_in, const int* in_sizes, int n_in,
                              void* d_out, int out_size, void* d_ws, size_t ws_size,
                              hipStream_t stream) {
    const float* mash_tags = (const float*)d_in[0];
    const float* targ_tags = (const float*)d_in[1];
    const float* cross     = (const float*)d_in[2];
    const float* E         = (const float*)d_in[3];
    const float* cand      = (const float*)d_in[4];
    const float* mdesc     = (const float*)d_in[5];
    const int*   clen      = (const int*)d_in[6];
    const float* WK        = (const float*)d_in[7];
    const float* WQ        = (const float*)d_in[8];
    const float* wide_w    = (const float*)d_in[9];
    const float* wide_b    = (const float*)d_in[10];
    const float* d1w       = (const float*)d_in[11];
    const float* d1b       = (const float*)d_in[12];
    const float* d2w       = (const float*)d_in[13];
    const float* d2b       = (const float*)d_in[14];
    const float* d3w       = (const float*)d_in[15];
    const float* d3b       = (const float*)d_in[16];
    const float* d4w       = (const float*)d_in[17];
    const float* d4b       = (const float*)d_in[18];
    const float* wwt       = (const float*)d_in[19];
    const float* dwt       = (const float*)d_in[20];
    const float* bias0     = (const float*)d_in[21];
    float* out = (float*)d_out;

    char* ws = (char*)d_ws;
    __bf16* din     = (__bf16*)(ws + 0);           // 8192*1536*2 = 25165824
    __bf16* h1      = (__bf16*)(ws + 25165824);    // 8192*512*2  =  8388608
    __bf16* h2      = (__bf16*)(ws + 33554432);    // 8192*256*2  =  4194304
    __bf16* h3      = (__bf16*)(ws + 37748736);    // 8192*128*2  =  2097152
    float*  widelin = (float*) (ws + 39845888);    // 8192*4      =    32768
    __bf16* w1bf    = (__bf16*)(ws + 39878656);    // 512*1536*2  =  1572864
    __bf16* w2bf    = (__bf16*)(ws + 41451520);    // 256*512*2   =   262144
    __bf16* w3bf    = (__bf16*)(ws + 41713664);    // 128*256*2   =    65536
    __bf16* w4bf    = (__bf16*)(ws + 41779200);    // 128*2

    // weight conversion (tiny)
    wd_cvt_bf16<<<(512 * 1536 + 255) / 256, 256, 0, stream>>>(d1w, w1bf, 512 * 1536);
    wd_cvt_bf16<<<(256 * 512  + 255) / 256, 256, 0, stream>>>(d2w, w2bf, 256 * 512);
    wd_cvt_bf16<<<(128 * 256  + 255) / 256, 256, 0, stream>>>(d3w, w3bf, 128 * 256);
    wd_cvt_bf16<<<1, 128, 0, stream>>>(d4w, w4bf, 128);

    // deep_in outer slices
    wd_pack_deep_in<<<(Bn * Dn + 255) / 256, 256, 0, stream>>>(mdesc, cand, din);

    // wide branch
    wd_wide<<<Bn, 256, 0, stream>>>(mash_tags, targ_tags, cross, wide_w, wide_b, widelin);

    // fused attention (dynamic LDS: 33920 floats = 135,680 B; 2 blocks/WGP)
    wd_attn<<<Bn, 256, 33920 * sizeof(float), stream>>>(E, cand, clen, WK, WQ, din);

    // deep MLP via bf16 WMMA (16x64 strip per wave, double-buffered k-loop)
    wd_gemm_bf16<<<(Bn / 16) * (512 / 64) / 8, 256, 0, stream>>>(din, w1bf, d1b, h1, Bn, 512, 1536, 1);
    wd_gemm_bf16<<<(Bn / 16) * (256 / 64) / 8, 256, 0, stream>>>(h1,  w2bf, d2b, h2, Bn, 256, 512, 1);
    wd_gemm_bf16<<<(Bn / 16) * (128 / 64) / 8, 256, 0, stream>>>(h2,  w3bf, d3b, h3, Bn, 128, 256, 1);

    // layer 4 + combine + sigmoid
    wd_final<<<(Bn + 255) / 256, 256, 0, stream>>>(h3, w4bf, d4b, widelin, wwt, dwt, bias0, out);
}
